// Model_67087389163600
// MI455X (gfx1250) — matrix operand order repaired and checked by reading, not compiled, not run on hardware
//
#include <hip/hip_runtime.h>
#include <hip/hip_bf16.h>
#include <stdint.h>

// Problem sizes (fixed by the reference)
#define E_N   200000
#define R_N   500
#define D2    512
#define B_N   512
#define KTILE 32
#define NKCH  (D2 / KTILE)      // 16 K-chunks
#define BLK_M 128
#define BLK_N 128
#define NT    1563              // ceil(200000/128)

typedef __attribute__((ext_vector_type(16))) __bf16 v16bf;
typedef __attribute__((ext_vector_type(8)))  float  v8f;

struct U32x8 { uint4 lo; uint4 hi; };
struct BF8   { __bf16 v[8]; };

static __device__ __forceinline__ v16bf bc16(uint4 a, uint4 b) {
  U32x8 t{a, b};
  return __builtin_bit_cast(v16bf, t);
}

// CDNA5 async copy: global -> LDS, 16B per lane, tracked by ASYNCcnt
static __device__ __forceinline__ void async_cp16(unsigned ldsByte, const void* gsrc) {
  asm volatile("global_load_async_to_lds_b128 %0, %1, off"
               :: "v"(ldsByte), "v"((unsigned long long)(uintptr_t)gsrc)
               : "memory");
}
static __device__ __forceinline__ void wait_async0() {
  asm volatile("s_wait_asynccnt 0" ::: "memory");
}

// ---------------------------------------------------------------------------
// 1) entity_emb f32 -> bf16 (one-time per call, ~205MB into workspace)
// ---------------------------------------------------------------------------
__global__ void k_convert(const float* __restrict__ src, __bf16* __restrict__ dst,
                          long long n) {
  long long i = ((long long)blockIdx.x * blockDim.x + threadIdx.x) * 8;
  if (i + 8 <= n) {
    float4 a = *(const float4*)(src + i);
    float4 b = *(const float4*)(src + i + 4);
    BF8 o;
    o.v[0] = (__bf16)a.x; o.v[1] = (__bf16)a.y; o.v[2] = (__bf16)a.z; o.v[3] = (__bf16)a.w;
    o.v[4] = (__bf16)b.x; o.v[5] = (__bf16)b.y; o.v[6] = (__bf16)b.z; o.v[7] = (__bf16)b.w;
    *(uint4*)(dst + i) = __builtin_bit_cast(uint4, o);
  }
}

// ---------------------------------------------------------------------------
// 2) build x = [h1*r1 - h2*r2, h2*r1 + h1*r2] (bf16), per-row factors, zero sums
// ---------------------------------------------------------------------------
__global__ void k_prep(const float* __restrict__ ent, const float* __restrict__ rel,
                       const float* __restrict__ weight, const float* __restrict__ degree,
                       const int* __restrict__ queries,
                       __bf16* __restrict__ Xbf, float* __restrict__ degv,
                       float* __restrict__ wfv, float* __restrict__ rowsum) {
  int b = blockIdx.x;          // query row
  int t = threadIdx.x;         // 0..255
  int head = queries[2 * b];
  int rl   = queries[2 * b + 1];
  const float* h = ent + (size_t)head * D2;
  const float* r = rel + (size_t)rl * D2;
  float h1 = h[t], h2 = h[t + 256];
  float r1 = r[t], r2 = r[t + 256];
  Xbf[(size_t)b * D2 + t]       = (__bf16)(h1 * r1 - h2 * r2);
  Xbf[(size_t)b * D2 + 256 + t] = (__bf16)(h2 * r1 + h1 * r2);
  if (t == 0) {
    degv[b]   = degree[(size_t)head * R_N + rl];
    wfv[b]    = fmaxf(1.0f + weight[(size_t)head * R_N + rl], 1e-5f);
    rowsum[b] = 0.0f;
  }
}

// ---------------------------------------------------------------------------
// 3) stable argsort(indexes) -> inverse permutation (tiny; counting sort)
// ---------------------------------------------------------------------------
__global__ void k_sort(const int* __restrict__ indexes, int* __restrict__ inv_order) {
  __shared__ int cnt[B_N + 1];
  __shared__ int ord[B_N];
  int t = threadIdx.x;
  for (int i = t; i < B_N + 1; i += blockDim.x) cnt[i] = 0;
  __syncthreads();
  if (t == 0) {
    for (int j = 0; j < B_N; ++j) cnt[indexes[j] + 1]++;
    for (int k = 0; k < B_N; ++k) cnt[k + 1] += cnt[k];
    for (int j = 0; j < B_N; ++j) { int key = indexes[j]; ord[cnt[key]++] = j; }
  }
  __syncthreads();
  for (int p = t; p < B_N; p += blockDim.x) inv_order[ord[p]] = p;
}

// ---------------------------------------------------------------------------
// 4) WMMA GEMM, async double-buffered B staging.
//    pass 0: rowsum += sum_e exp(logit). pass 1: write normalized, scaled,
//    clamped scores to out[inv_order[m]].
//    Block: 256 thr = 8 waves; tile 128M x 128N; K chunks of 32 (bf16 WMMA).
// ---------------------------------------------------------------------------
__global__ __launch_bounds__(256)
void k_gemm(const __bf16* __restrict__ Ebf, const __bf16* __restrict__ Xbf,
            const float* __restrict__ rowsum_in, const float* __restrict__ degv,
            const float* __restrict__ wfv, const int* __restrict__ inv_order,
            float* __restrict__ rowsum_out, float* __restrict__ out, int pass) {
  // Double-buffered B tile in WMMA B-fragment layout:
  // 2 x (8 subtiles x 32 lanes x 32B) = 16KB
  __shared__ uint4 smB[1024];

  const int tid   = threadIdx.x;
  const int lane  = tid & 31;
  const int wave  = tid >> 5;          // M subtile 0..7
  const int lhalf = lane >> 4;         // 0: lanes 0-15, 1: lanes 16-31
  const int lmod  = lane & 15;
  const int e0 = blockIdx.x * BLK_N;
  const int m0 = blockIdx.y * BLK_M;

  v8f acc[8] = {};

  // cooperative B loader: thread -> (row-in-tile, K half), 32 bytes each
  const int ln    = tid >> 1;          // 0..127 : N row within tile
  const int lpart = tid & 1;           // 0: K 0..15 (lanes 0-15), 1: K 16..31
  int grow = e0 + ln; if (grow >= E_N) grow = E_N - 1;
  const __bf16* gB = Ebf + (size_t)grow * D2 + lpart * 16;
  // byte offset of this thread's 32B slot inside one 8KB tile buffer
  const unsigned slotByte =
      (unsigned)((((ln >> 4) * 32) + (ln & 15) + lpart * 16) * 32);
  const unsigned ldsBase = (unsigned)(uintptr_t)smB;

  // A fragment source (16x32 bf16 WMMA A layout): two 16B chunks per lane
  const __bf16* gA = Xbf + (size_t)(m0 + wave * 16 + lmod) * D2 + lhalf * 16;

  // prologue: stage tile 0 into buffer 0
  {
    unsigned d = ldsBase + slotByte;
    async_cp16(d,      gB);
    async_cp16(d + 16, gB + 8);
  }

  for (int kc = 0; kc < NKCH; ++kc) {
    const int k0 = kc * KTILE;
    // A fragment (L2-resident; plain loads)
    uint4 a0 = *(const uint4*)(gA + k0);
    uint4 a1 = *(const uint4*)(gA + k0 + 8);

    wait_async0();        // own async stores for tile kc are in LDS
    __syncthreads();      // everyone's tile kc staged; prev buffer fully read

    if (kc + 1 < NKCH) {  // stream tile kc+1 into the other buffer
      unsigned d = ldsBase + ((kc + 1) & 1) * 8192 + slotByte;
      const __bf16* s = gB + (size_t)(kc + 1) * KTILE;
      async_cp16(d,      s);
      async_cp16(d + 16, s + 8);
    }

    v16bf af = bc16(a0, a1);
    const uint4* bufp = &smB[(kc & 1) * 512];
#pragma unroll
    for (int s = 0; s < 8; ++s) {
      const uint4* bp = &bufp[(s * 32 + lane) * 2];
      v16bf bf = bc16(bp[0], bp[1]);
      acc[s] = __builtin_amdgcn_wmma_f32_16x16x32_bf16(
          false, af, false, bf, (short)0, acc[s], false, false);
    }
    // schedule: 2 DS reads then 1 WMMA, repeated -> only ~2 B frags live,
    // keeps accumulators pinned (no v_mov shuffles, no WMMA->VALU nops)
#pragma unroll
    for (int s = 0; s < 8; ++s) {
      __builtin_amdgcn_sched_group_barrier(0x100, 2, 0); // DS read x2
      __builtin_amdgcn_sched_group_barrier(0x008, 1, 0); // WMMA x1
    }
  }

  // C layout: element (vgpr v, lane) = C[M = v + 8*lhalf][N = lmod] per subtile
  const int mrow = m0 + wave * 16;
  if (pass == 0) {
    float rs[8];
#pragma unroll
    for (int v = 0; v < 8; ++v) rs[v] = 0.0f;
#pragma unroll
    for (int s = 0; s < 8; ++s) {
      int e = e0 + s * 16 + lmod;
      bool ok = (e < E_N);
#pragma unroll
      for (int v = 0; v < 8; ++v) {
        float x = __expf(acc[s][v]);           // logits ~1e-4: no max needed
        rs[v] += ok ? x : 0.0f;
      }
    }
#pragma unroll
    for (int v = 0; v < 8; ++v) {
      float x = rs[v];
#pragma unroll
      for (int off = 1; off < 16; off <<= 1) x += __shfl_xor(x, off, 32);
      if (lmod == 0) atomicAdd(&rowsum_out[mrow + v + 8 * lhalf], x);
    }
  } else {
#pragma unroll
    for (int v = 0; v < 8; ++v) {
      int m = mrow + v + 8 * lhalf;
      float inv = 1.0f / rowsum_in[m];
      float dg = degv[m], wf = wfv[m];
      size_t orow = (size_t)inv_order[m] * (size_t)E_N;
#pragma unroll
      for (int s = 0; s < 8; ++s) {
        int e = e0 + s * 16 + lmod;
        if (e < E_N) {
          float sc = __expf(acc[s][v]) * inv;
          sc = fminf(sc * dg, 1.0f);
          sc = fminf(sc * wf, 1.0f);
          out[orow + e] = sc;
        }
      }
    }
  }
}

// ---------------------------------------------------------------------------
extern "C" void kernel_launch(void* const* d_in, const int* in_sizes, int n_in,
                              void* d_out, int out_size, void* d_ws, size_t ws_size,
                              hipStream_t stream) {
  const float* ent    = (const float*)d_in[0];
  const float* rel    = (const float*)d_in[1];
  const float* weight = (const float*)d_in[2];
  const float* degree = (const float*)d_in[3];
  const int*   quer   = (const int*)d_in[4];
  const int*   idx    = (const int*)d_in[5];
  float* out = (float*)d_out;

  // workspace layout (~205.3 MB)
  char* ws = (char*)d_ws;
  __bf16* Ebf = (__bf16*)ws;                                   // 204,800,000 B
  __bf16* Xbf = (__bf16*)(ws + 204800000);                     // 524,288 B
  float*  degv = (float*)(ws + 204800000 + 524288);            // 2 KB
  float*  wfv  = degv + B_N;
  float*  rowsum = wfv + B_N;
  int*    inv_order = (int*)(rowsum + B_N);

  k_convert<<<50000, 256, 0, stream>>>(ent, Ebf, (long long)E_N * D2);
  k_prep<<<B_N, 256, 0, stream>>>(ent, rel, weight, degree, quer, Xbf, degv, wfv, rowsum);
  k_sort<<<1, 256, 0, stream>>>(idx, inv_order);

  dim3 grid(NT, B_N / BLK_M);   // 1563 x 4
  k_gemm<<<grid, 256, 0, stream>>>(Ebf, Xbf, rowsum, degv, wfv, inv_order, rowsum, out, 0);
  k_gemm<<<grid, 256, 0, stream>>>(Ebf, Xbf, rowsum, degv, wfv, inv_order, rowsum, out, 1);
}
